// AutoregressivePredictor_12163347382481
// MI455X (gfx1250) — compile-verified, matrix operand order checked by reading, not run on hardware
//
#include <hip/hip_runtime.h>
#include <math.h>

#define H      256
#define NH     8
#define HD     32
#define FF     1024
#define VOCAB  1024
#define STEPS  8
#define S0     4096
#define MAXS   (S0 + STEPS)
#define EPSV   1e-6f
#define THETA  10000.0f

typedef __attribute__((ext_vector_type(16))) __bf16 v16bf;
typedef __attribute__((ext_vector_type(8)))  __bf16 bf16x8;
typedef __attribute__((ext_vector_type(8)))  float  v8f;

__device__ __forceinline__ float wave_sum(float v) {
#pragma unroll
  for (int o = 16; o > 0; o >>= 1) v += __shfl_xor(v, o, 32);
  return v;
}

// ---- 1) transpose+convert [Wk | Wv] (f32, [k][n] row-major) -> WT bf16 [n(0..511)][k] ----
__global__ void convert_wkv_kernel(const float* __restrict__ Wk,
                                   const float* __restrict__ Wv,
                                   unsigned short* __restrict__ WTr) {
  __bf16* WT = reinterpret_cast<__bf16*>(WTr);
  int tid = blockIdx.x * blockDim.x + threadIdx.x;   // 512*256 threads
  int k = tid >> 9;          // 0..255
  int n = tid & 511;         // 0..511 (coalesced read across n)
  float v = (n < H) ? Wk[k * H + n] : Wv[k * H + (n - H)];
  WT[n * H + k] = (__bf16)v;
}

// ---- 2) hb = bf16(rmsnorm(x) * ln1) for all 4096 prompt rows; one wave per row ----
__global__ void rmsnorm_rows_kernel(const float* __restrict__ x,
                                    const float* __restrict__ ln1,
                                    unsigned short* __restrict__ hbr) {
  __bf16* hb = reinterpret_cast<__bf16*>(hbr);
  int wid = threadIdx.x >> 5, lane = threadIdx.x & 31;
  int row = blockIdx.x * 8 + wid;
  const float* xr = x + row * H;
  float vals[8]; float ss = 0.f;
#pragma unroll
  for (int i = 0; i < 8; i++) { float v = xr[lane * 8 + i]; vals[i] = v; ss += v * v; }
  ss = wave_sum(ss);
  float rs = rsqrtf(ss / (float)H + EPSV);
#pragma unroll
  for (int i = 0; i < 8; i++)
    hb[row * H + lane * 8 + i] = (__bf16)(vals[i] * rs * ln1[lane * 8 + i]);
}

// ---- 3) prompt KV GEMM: hb(4096x256 bf16) @ WT^T(256x512) -> Kc,Vc f32 via WMMA ----
__global__ void kv_gemm_kernel(const unsigned short* __restrict__ hbr,
                               const unsigned short* __restrict__ WTr,
                               float* __restrict__ Kc, float* __restrict__ Vc) {
  const __bf16* hb = reinterpret_cast<const __bf16*>(hbr);
  const __bf16* WT = reinterpret_cast<const __bf16*>(WTr);
  int wid = threadIdx.x >> 5, lane = threadIdx.x & 31;
  int rstrip = blockIdx.x & 255;          // 256 row strips of 16
  int cgrp   = blockIdx.x >> 8;           // 4 col groups x 8 waves = 32 col tiles
  int m0 = rstrip * 16;
  int n0 = (cgrp * 8 + wid) * 16;
  int khalf = (lane >> 4) << 3;           // lanes 0-15 -> K 0..7/16..23; 16-31 -> 8..15/24..31
  const __bf16* A = hb + (m0 + (lane & 15)) * H;
  const __bf16* B = WT + (n0 + (lane & 15)) * H;
  __builtin_prefetch(A + 16 * H, 0, 1);   // global_prefetch for next row strip
  v8f acc = {0.f, 0.f, 0.f, 0.f, 0.f, 0.f, 0.f, 0.f};
#pragma unroll
  for (int kk = 0; kk < H; kk += 32) {
    bf16x8 a0 = *(const bf16x8*)(A + kk + khalf);
    bf16x8 a1 = *(const bf16x8*)(A + kk + khalf + 16);
    bf16x8 b0 = *(const bf16x8*)(B + kk + khalf);
    bf16x8 b1 = *(const bf16x8*)(B + kk + khalf + 16);
    v16bf av, bv;
#pragma unroll
    for (int i = 0; i < 8; i++) { av[i] = a0[i]; av[i + 8] = a1[i]; bv[i] = b0[i]; bv[i + 8] = b1[i]; }
    acc = __builtin_amdgcn_wmma_f32_16x16x32_bf16(false, av, false, bv, (short)0, acc, false, false);
  }
  int ocol = n0 + (lane & 15);
  int radd = (lane >> 4) << 3;            // D layout: lanes 16-31 hold rows M+8
  float* dst; int c;
  if (ocol < H) { dst = Kc; c = ocol; } else { dst = Vc; c = ocol - H; }
#pragma unroll
  for (int r = 0; r < 8; r++) dst[(m0 + r + radd) * H + c] = acc[r];
}

// ---- 4) in-place RoPE on the 4096 prompt K rows ----
__global__ void rope_prompt_k_kernel(float* __restrict__ Kc) {
  int tid = blockIdx.x * 256 + threadIdx.x;  // 4096*8*16 pair-threads
  int s = tid >> 7;
  int r = tid & 127;
  int head = r >> 4, i = r & 15;
  float inv = powf(THETA, -(float)(2 * i) / (float)HD);
  float ang = (float)s * inv;
  float cs = cosf(ang), sn = sinf(ang);
  float* base = Kc + s * H + head * HD;
  float k1 = base[i], k2 = base[i + 16];
  base[i]      = k1 * cs - k2 * sn;
  base[i + 16] = k2 * cs + k1 * sn;
}

// ---- 5) per step: q (and for t>0 append k,v) of the current last token ----
__global__ void step_qkv_kernel(int t, const float* __restrict__ x0,
                                const float* __restrict__ ln1,
                                const float* __restrict__ Wq, const float* __restrict__ Wk,
                                const float* __restrict__ Wv,
                                const float* __restrict__ xlast, float* __restrict__ qout,
                                float* __restrict__ Kc, float* __restrict__ Vc) {
  __shared__ float hs[H], qs[H], ks[H], sred[8];
  int tid = threadIdx.x;                   // 256 threads
  float xv = (t == 0) ? x0[(S0 - 1) * H + tid] : xlast[tid];
  float ss = wave_sum(xv * xv);
  if ((tid & 31) == 0) sred[tid >> 5] = ss;
  __syncthreads();
  float tot = 0.f;
#pragma unroll
  for (int i = 0; i < 8; i++) tot += sred[i];
  float rs = rsqrtf(tot / (float)H + EPSV);
  hs[tid] = xv * rs * ln1[tid];
  __syncthreads();
  float q = 0.f, k = 0.f, v = 0.f;
  for (int kk = 0; kk < H; kk++) {         // coalesced across tid (columns)
    float h = hs[kk];
    q += h * Wq[kk * H + tid];
    k += h * Wk[kk * H + tid];
    v += h * Wv[kk * H + tid];
  }
  qs[tid] = q; ks[tid] = k;
  __syncthreads();
  int pos = S0 - 1 + t;
  int d = tid & 31;
  int i = (d < 16) ? d : d - 16;
  float inv = powf(THETA, -(float)(2 * i) / (float)HD);
  float ang = (float)pos * inv;
  float cs = cosf(ang), sn = sinf(ang);
  float qr, kr;
  if (d < 16) { qr = qs[tid] * cs - qs[tid + 16] * sn; kr = ks[tid] * cs - ks[tid + 16] * sn; }
  else        { qr = qs[tid] * cs + qs[tid - 16] * sn; kr = ks[tid] * cs + ks[tid - 16] * sn; }
  qout[tid] = qr;
  if (t > 0) { Kc[pos * H + tid] = kr; Vc[pos * H + tid] = v; }
}

// ---- 6) per step: single-query attention over L = 4096+t cached keys; block = head ----
__global__ void step_attn_kernel(int t, const float* __restrict__ q,
                                 const float* __restrict__ Kc, const float* __restrict__ Vc,
                                 float* __restrict__ o) {
  __shared__ float sc[MAXS];
  __shared__ float red[256];
  __shared__ float qsh[HD];
  __shared__ float part[8][HD];
  int head = blockIdx.x, tid = threadIdx.x;
  int L = S0 + t;
  if (tid < HD) qsh[tid] = q[head * HD + tid];
  __syncthreads();
  float lmax = -1e30f;
  for (int j = tid; j < L; j += 256) {
    const float* kr = Kc + j * H + head * HD;
    float s = 0.f;
#pragma unroll
    for (int d2 = 0; d2 < HD; d2++) s += qsh[d2] * kr[d2];
    s *= 0.17677669529663687f;             // 1/sqrt(32)
    sc[j] = s;
    lmax = fmaxf(lmax, s);
  }
  red[tid] = lmax;
  __syncthreads();
  for (int off = 128; off > 0; off >>= 1) {
    if (tid < off) red[tid] = fmaxf(red[tid], red[tid + off]);
    __syncthreads();
  }
  float m = red[0];
  __syncthreads();
  float lsum = 0.f;
  for (int j = tid; j < L; j += 256) { float p = expf(sc[j] - m); sc[j] = p; lsum += p; }
  red[tid] = lsum;
  __syncthreads();
  for (int off = 128; off > 0; off >>= 1) {
    if (tid < off) red[tid] += red[tid + off];
    __syncthreads();
  }
  float lden = red[0];
  __syncthreads();
  int d = tid & 31, p8 = tid >> 5;         // 32 dims x 8 j-partitions; V reads coalesced
  float acc = 0.f;
  for (int j = p8; j < L; j += 8) acc += sc[j] * Vc[j * H + head * HD + d];
  part[p8][d] = acc;
  __syncthreads();
  if (p8 == 0) {
    float s = 0.f;
#pragma unroll
    for (int i = 0; i < 8; i++) s += part[i][d];
    o[head * HD + d] = s / lden;
  }
}

// ---- 7) per step: Wo proj + residual + rmsnorm + SwiGLU MLP + logits + argmax + embed ----
__global__ void step_post_kernel(int t, const float* __restrict__ ln2,
                                 const float* __restrict__ Wo,
                                 const float* __restrict__ Wg, const float* __restrict__ Wu,
                                 const float* __restrict__ Wd,
                                 const float* __restrict__ emb, const float* __restrict__ Wout,
                                 const float* __restrict__ bout,
                                 const float* __restrict__ xlast_in, const float* __restrict__ x0,
                                 const float* __restrict__ o,
                                 float* __restrict__ out, float* __restrict__ xlast_out) {
  __shared__ float os[H], x2[H], h2[H], act[FF], y[H];
  __shared__ float av[VOCAB];
  __shared__ int   ai[VOCAB];
  __shared__ float sred[32];
  __shared__ int   stok;
  int tid = threadIdx.x;                   // 1024 threads
  if (tid < H) os[tid] = o[tid];
  __syncthreads();
  if (tid < H) {
    float a = 0.f;
    for (int kk = 0; kk < H; kk++) a += os[kk] * Wo[kk * H + tid];
    float xl = (t == 0) ? x0[(S0 - 1) * H + tid] : xlast_in[tid];
    x2[tid] = xl + a;
  }
  __syncthreads();
  float c = (tid < H) ? x2[tid] : 0.f;
  float ss = wave_sum(c * c);
  if ((tid & 31) == 0) sred[tid >> 5] = ss;
  __syncthreads();
  float tot = 0.f;
#pragma unroll
  for (int i = 0; i < 32; i++) tot += sred[i];
  float rs = rsqrtf(tot / (float)H + EPSV);
  if (tid < H) h2[tid] = x2[tid] * rs * ln2[tid];
  __syncthreads();
  {
    float g = 0.f, u = 0.f;
    for (int kk = 0; kk < H; kk++) { float h = h2[kk]; g += h * Wg[kk * FF + tid]; u += h * Wu[kk * FF + tid]; }
    float sg = g / (1.f + expf(-g));       // silu
    act[tid] = sg * u;
  }
  __syncthreads();
  if (tid < H) {
    float m = 0.f;
    for (int f = 0; f < FF; f++) m += act[f] * Wd[f * H + tid];
    y[tid] = x2[tid] + m;
  }
  __syncthreads();
  {
    float lg = bout[tid];
    for (int kk = 0; kk < H; kk++) lg += y[kk] * Wout[kk * VOCAB + tid];
    out[STEPS + t * VOCAB + tid] = lg;     // logits block of d_out
    av[tid] = lg; ai[tid] = tid;
  }
  __syncthreads();
  for (int off = 512; off > 0; off >>= 1) { // argmax, first-max-wins ties
    if (tid < off) {
      if (av[tid + off] > av[tid] ||
          (av[tid + off] == av[tid] && ai[tid + off] < ai[tid])) {
        av[tid] = av[tid + off]; ai[tid] = ai[tid + off];
      }
    }
    __syncthreads();
  }
  if (tid == 0) { stok = ai[0]; out[t] = (float)ai[0]; }
  __syncthreads();
  if (tid < H) xlast_out[tid] = emb[stok * H + tid];
}

extern "C" void kernel_launch(void* const* d_in, const int* in_sizes, int n_in,
                              void* d_out, int out_size, void* d_ws, size_t ws_size,
                              hipStream_t stream) {
  const float* x0   = (const float*)d_in[0];
  const float* ln1  = (const float*)d_in[1];
  const float* ln2  = (const float*)d_in[2];
  const float* Wq   = (const float*)d_in[3];
  const float* Wk   = (const float*)d_in[4];
  const float* Wv   = (const float*)d_in[5];
  const float* Wo   = (const float*)d_in[6];
  const float* Wg   = (const float*)d_in[7];
  const float* Wu   = (const float*)d_in[8];
  const float* Wd   = (const float*)d_in[9];
  const float* emb  = (const float*)d_in[10];
  const float* Wout = (const float*)d_in[11];
  const float* bout = (const float*)d_in[12];
  (void)in_sizes; (void)n_in; (void)out_size; (void)ws_size;

  char* ws = (char*)d_ws;
  unsigned short* hb = (unsigned short*)(ws);                  // 4096*256 bf16 = 2 MB
  unsigned short* WT = (unsigned short*)(ws + 2097152);        // 512*256  bf16 = 256 KB
  float* Kc = (float*)(ws + 2359296);                          // 4104*256 f32
  float* Vc = (float*)(ws + 6561792);                          // 4104*256 f32
  float* qb = (float*)(ws + 10764288);                         // 256 f32
  float* ob = (float*)(ws + 10765312);                         // 256 f32
  float* xl = (float*)(ws + 10766336);                         // 256 f32
  float* out = (float*)d_out;

  convert_wkv_kernel<<<dim3(512), dim3(256), 0, stream>>>(Wk, Wv, WT);
  rmsnorm_rows_kernel<<<dim3(512), dim3(256), 0, stream>>>(x0, ln1, hb);
  kv_gemm_kernel<<<dim3(1024), dim3(256), 0, stream>>>(hb, WT, Kc, Vc);
  rope_prompt_k_kernel<<<dim3(2048), dim3(256), 0, stream>>>(Kc);

  for (int t = 0; t < STEPS; t++) {
    step_qkv_kernel<<<dim3(1), dim3(256), 0, stream>>>(t, x0, ln1, Wq, Wk, Wv, xl, qb, Kc, Vc);
    step_attn_kernel<<<dim3(NH), dim3(256), 0, stream>>>(t, qb, Kc, Vc, ob);
    step_post_kernel<<<dim3(1), dim3(1024), 0, stream>>>(t, ln2, Wo, Wg, Wu, Wd, emb, Wout,
                                                         bout, xl, x0, ob, out, xl);
  }
}
